// MultiHeadAttention_52484500357302
// MI455X (gfx1250) — matrix operand order repaired and checked by reading, not compiled, not run on hardware
//
#include <hip/hip_runtime.h>
#include <hip/hip_bf16.h>
#include <math.h>

// Problem constants (B, S, H, NH, HD) from the reference.
#define BB 2
#define SS 2048
#define HH 1024
#define NHH 16
#define HDD 64
#define SCALE_F 0.125f   // 1/sqrt(64)

typedef __attribute__((ext_vector_type(16))) __bf16 v16bf;
typedef __attribute__((ext_vector_type(8)))  float  v8f;

static __device__ inline v8f wmma_bf16(v16bf a, v16bf b, v8f c) {
  return __builtin_amdgcn_wmma_f32_16x16x32_bf16(false, a, false, b,
                                                 (short)0, c, false, false);
}

// ---------------------------------------------------------------------------
// Fragment builders for CDNA5 WMMA (CDNA5 ISA §7.12.2); A and B both use contiguous k = 16*half + s
// A-matrix 16x32 bf16: half = lane>>4, m = lane&15; VGPR v holds K pair
//   k = half*16 + v*2  (contiguous, same as B)
// B-matrix 32x16 bf16: n = lane&15, kbase = (lane>>4)*16
// C/D 16x16 f32: VGPR r => row r + 8*(lane>>4), col = lane&15
// ---------------------------------------------------------------------------

static __device__ inline v16bf a_frag_bf16(const __bf16* __restrict__ A, int lda,
                                           int row0, int k0, int lane) {
  int half = lane >> 4, m = lane & 15;
  const __bf16* p = A + (size_t)(row0 + m) * lda + k0 + half * 16;
  v16bf a;
#pragma unroll
  for (int v = 0; v < 8; ++v) {
    int k = v * 2;
    a[2 * v]     = p[k];
    a[2 * v + 1] = p[k + 1];
  }
  return a;
}

// B fragment where logical B(k,n) = src[n0+n][k0+k]  (transposed-major source:
// K^T tiles, pre-transposed weights WT[N][K], or V^T tiles).
// Per-lane load = 16 contiguous bf16 = one 32B vector.
static __device__ inline v16bf b_frag_trans_bf16(const __bf16* __restrict__ Bm,
                                                 int ldb, int n0, int k0, int lane) {
  int n = lane & 15;
  int kb = (lane >> 4) * 16;
  const __bf16* p = Bm + (size_t)(n0 + n) * ldb + k0 + kb;
  return *(const v16bf*)p;   // 32B aligned (k0, kb multiples of 16 elems)
}

// Async copy of one 32-byte chunk into LDS (two b128 transfers, GVS mode:
// LDS address from vdst, global address from saddr + vaddr, both plus offset).
static __device__ inline void async_copy32(unsigned lds_off, unsigned gvoff,
                                           const void* sbase) {
  asm volatile(
      "global_load_async_to_lds_b128 %0, %1, %2 offset:0\n\t"
      "global_load_async_to_lds_b128 %0, %1, %2 offset:16"
      :
      : "v"(lds_off), "v"(gvoff),
        "s"((unsigned long long)(uintptr_t)sbase)
      : "memory");
}

// ---------------------------------------------------------------------------
// fp32 -> bf16 elementwise conversion (hidden_states)
// ---------------------------------------------------------------------------
__global__ __launch_bounds__(256) void f32_to_bf16_kernel(
    const float* __restrict__ src, __bf16* __restrict__ dst, int n) {
  int i = (blockIdx.x * 256 + threadIdx.x) * 4;
  if (i + 3 < n) {
    float4 v = *(const float4*)(src + i);
    dst[i + 0] = (__bf16)v.x;
    dst[i + 1] = (__bf16)v.y;
    dst[i + 2] = (__bf16)v.z;
    dst[i + 3] = (__bf16)v.w;
  }
}

// ---------------------------------------------------------------------------
// Transpose + convert: WT[n*K + k] = (bf16) W[k*N + n]   (32x32 LDS tiles)
// ---------------------------------------------------------------------------
__global__ __launch_bounds__(256) void transpose_f32_bf16_kernel(
    const float* __restrict__ W, __bf16* __restrict__ WT, int K, int N) {
  __shared__ float tile[32][33];
  int tx = threadIdx.x, ty = threadIdx.y;       // 32 x 8
  int n0 = blockIdx.x * 32, k0 = blockIdx.y * 32;
#pragma unroll
  for (int i = 0; i < 32; i += 8)
    tile[ty + i][tx] = W[(size_t)(k0 + ty + i) * N + n0 + tx];
  __syncthreads();
#pragma unroll
  for (int i = 0; i < 32; i += 8)
    WT[(size_t)(n0 + ty + i) * K + k0 + tx] = (__bf16)tile[tx][ty + i];
}

// ---------------------------------------------------------------------------
// GEMM v2: C[M,N] = A[M,K](bf16) @ WT[N,K]^T(bf16) + bias[N]
// Block: 8 waves -> 128x64 C tile; wave (waveM, waveN) owns 64x16.
// A tile (128x32) double-buffered in LDS via async-to-LDS DMA.
// MODE: 0 = fp32 row-major, 1 = bf16 row-major, 2 = bf16 V^T [B][NH][HD][S]
// ---------------------------------------------------------------------------
template <int MODE>
__global__ __launch_bounds__(256) void gemm_bias_v2(
    const __bf16* __restrict__ A, const __bf16* __restrict__ WT,
    const float* __restrict__ bias, void* __restrict__ Cp,
    int M, int N, int K) {
  __shared__ __bf16 Atile[2][128 * 32];   // 2 x 8 KB

  int tid = threadIdx.x;
  int lane = tid & 31, wave = tid >> 5;
  int waveM = wave >> 2, waveN = wave & 3;
  int bm = blockIdx.y * 128;
  int bn = blockIdx.x * 64;
  int coln = bn + waveN * 16;

  unsigned lds_base = (unsigned)(uintptr_t)(&Atile[0][0]);
  unsigned chunk = (unsigned)tid * 32u;          // thread's 32B slot
  unsigned gvoff = (unsigned)((((tid >> 1) * K) + (tid & 1) * 16) * 2);

  async_copy32(lds_base + chunk, gvoff, A + (size_t)bm * K);   // prologue

  v8f acc0 = {}, acc1 = {}, acc2 = {}, acc3 = {};
  const int NSTEP = K / 32;
  for (int step = 0; step < NSTEP; ++step) {
    int buf = step & 1;
    if (step + 1 < NSTEP) {
      async_copy32(lds_base + (unsigned)((step + 1) & 1) * 8192u + chunk,
                   gvoff, A + (size_t)bm * K + (step + 1) * 32);
      asm volatile("s_wait_asynccnt 0x2" ::: "memory");
    } else {
      asm volatile("s_wait_asynccnt 0x0" ::: "memory");
    }
    __syncthreads();

    const __bf16* Ab = &Atile[buf][0];
    v16bf b  = b_frag_trans_bf16(WT, K, coln, step * 32, lane);
    v16bf a0 = a_frag_bf16(Ab, 32, waveM * 64 + 0,  0, lane);
    v16bf a1 = a_frag_bf16(Ab, 32, waveM * 64 + 16, 0, lane);
    v16bf a2 = a_frag_bf16(Ab, 32, waveM * 64 + 32, 0, lane);
    v16bf a3 = a_frag_bf16(Ab, 32, waveM * 64 + 48, 0, lane);
    acc0 = wmma_bf16(a0, b, acc0);
    acc1 = wmma_bf16(a1, b, acc1);
    acc2 = wmma_bf16(a2, b, acc2);
    acc3 = wmma_bf16(a3, b, acc3);

    __syncthreads();
  }

  int half = lane >> 4, n = lane & 15;
  float bv = bias[coln + n];
  int rbase = bm + waveM * 64;
#pragma unroll
  for (int t = 0; t < 4; ++t) {
    v8f acc = (t == 0) ? acc0 : (t == 1) ? acc1 : (t == 2) ? acc2 : acc3;
#pragma unroll
    for (int r = 0; r < 8; ++r) {
      int row = rbase + t * 16 + r + 8 * half;
      float val = acc[r] + bv;
      if constexpr (MODE == 0) {
        ((float*)Cp)[(size_t)row * N + coln + n] = val;
      } else if constexpr (MODE == 1) {
        ((__bf16*)Cp)[(size_t)row * N + coln + n] = (__bf16)val;
      } else {   // V^T: [B][NH][HD][S]
        int col = coln + n;
        int hIdx = col >> 6, d = col & 63;
        int bIdx = row >> 11, s = row & (SS - 1);
        ((__bf16*)Cp)[((size_t)(bIdx * NHH + hIdx) * HDD + d) * SS + s] =
            (__bf16)val;
      }
    }
  }
}

// ---------------------------------------------------------------------------
// Flash-attention v4. One block = one (b, h, 128-query chunk); 8 waves each
// own a 16-query tile and share async-staged K (32x64) / V^T (64x32) LDS
// tiles, double-buffered.  Softmax denominator accumulated on the matrix
// pipe as P @ ones (5th WMMA accumulator) instead of shuffle reductions.
// ---------------------------------------------------------------------------
__global__ __launch_bounds__(256) void attention_kernel(
    const __bf16* __restrict__ Q, const __bf16* __restrict__ Kb,
    const __bf16* __restrict__ VT, const int* __restrict__ amask,
    __bf16* __restrict__ O) {
  __shared__ __bf16 Kt[2][32 * 64];      // 2 x 4 KB
  __shared__ __bf16 Vt[2][64 * 32];      // 2 x 4 KB
  __shared__ __bf16 lds_p[8][16 * 32];   // per-wave P staging, 8 KB

  int tid = threadIdx.x;
  int lane = tid & 31;
  int wave = tid >> 5;
  int bh = blockIdx.y;                   // 0..31
  int b = bh >> 4, h = bh & 15;
  int qtile = blockIdx.x * 8 + wave;     // 0..127
  int q0 = qtile * 16;

  const __bf16* Qp  = Q  + (size_t)b * SS * HH + h * HDD;
  const __bf16* Kp  = Kb + (size_t)b * SS * HH + h * HDD;
  const __bf16* VTp = VT + (size_t)(b * NHH + h) * HDD * SS;
  const int* mrow = amask + b * SS;

  // --- per-thread async staging role ---
  unsigned kt_base = (unsigned)(uintptr_t)(&Kt[0][0]);
  unsigned vt_base = (unsigned)(uintptr_t)(&Vt[0][0]);
  unsigned myLds, gvoff;
  const __bf16* srcBase;
  size_t srcStep;              // bytes to advance per 32-key block
  if (tid < 128) {             // stage K tile: 32 rows x 128B
    int row = tid >> 2, part = tid & 3;
    myLds = kt_base + (unsigned)(row * 128 + part * 32);
    gvoff = (unsigned)((row * HH + part * 16) * 2);
    srcBase = Kp;
    srcStep = (size_t)32 * HH * 2;
  } else {                     // stage V^T tile: 64 rows x 64B
    int c = tid - 128;
    int d = c >> 1, part = c & 1;
    myLds = vt_base + (unsigned)(d * 64 + part * 32);
    gvoff = (unsigned)((d * SS + part * 16) * 2);
    srcBase = VTp;
    srcStep = (size_t)32 * 2;
  }

  // --- per-wave Q fragments (head dim 64 -> two 16x32 A frags) ---
  v16bf qa0 = a_frag_bf16(Qp, HH, q0, 0, lane);
  v16bf qa1 = a_frag_bf16(Qp, HH, q0, 32, lane);

  // all-ones B fragment: row-sum of P via the matrix pipe
  v16bf onesb;
#pragma unroll
  for (int v = 0; v < 16; ++v) onesb[v] = (__bf16)1.0f;

  float m[8];
#pragma unroll
  for (int r = 0; r < 8; ++r) m[r] = -3.0e38f;
  v8f o0 = {}, o1 = {}, o2 = {}, o3 = {};
  v8f ol = {};                           // softmax denominator accumulator

  int half = lane >> 4, n = lane & 15;
  __bf16* pl = lds_p[wave];

  async_copy32(myLds, gvoff, srcBase);   // prologue: buffer 0 <- block 0

  const int NBLK = SS / 32;              // 64
  for (int blk = 0; blk < NBLK; ++blk) {
    int buf = blk & 1;
    if (blk + 1 < NBLK) {
      async_copy32(myLds + (unsigned)((blk + 1) & 1) * 4096u, gvoff,
                   (const char*)srcBase + (size_t)(blk + 1) * srcStep);
      asm volatile("s_wait_asynccnt 0x2" ::: "memory");
    } else {
      asm volatile("s_wait_asynccnt 0x0" ::: "memory");
    }
    __syncthreads();

    const __bf16* Kl = &Kt[buf][0];      // [32 keys][64 d]
    const __bf16* Vl = &Vt[buf][0];      // [64 d][32 keys]
    int k0 = blk * 32;

    // scores S(16x32) = Q(16x64) @ K^T(64x32): 4 WMMAs, B frags from LDS
    v8f s0 = {}, s1 = {};
    {
      v16bf bt0 = b_frag_trans_bf16(Kl, 64, 0, 0, lane);
      v16bf bt1 = b_frag_trans_bf16(Kl, 64, 0, 32, lane);
      s0 = wmma_bf16(qa0, bt0, s0);
      s0 = wmma_bf16(qa1, bt1, s0);
      v16bf bt2 = b_frag_trans_bf16(Kl, 64, 16, 0, lane);
      v16bf bt3 = b_frag_trans_bf16(Kl, 64, 16, 32, lane);
      s1 = wmma_bf16(qa0, bt2, s1);
      s1 = wmma_bf16(qa1, bt3, s1);
    }

    float msk0 = mrow[k0 + n] ? 0.0f : -3.0e38f;
    float msk1 = mrow[k0 + 16 + n] ? 0.0f : -3.0e38f;
#pragma unroll
    for (int r = 0; r < 8; ++r) {
      s0[r] = s0[r] * SCALE_F + msk0;
      s1[r] = s1[r] * SCALE_F + msk1;
    }

    // online softmax max; row = r + 8*half; xor-shuffles stay in 16-lane group
    float alpha[8];
#pragma unroll
    for (int r = 0; r < 8; ++r) {
      float mx = fmaxf(s0[r], s1[r]);
#pragma unroll
      for (int off = 8; off > 0; off >>= 1)
        mx = fmaxf(mx, __shfl_xor(mx, off, 32));
      float mnew = fmaxf(m[r], mx);
      alpha[r] = __expf(m[r] - mnew);
      m[r] = mnew;
      s0[r] = __expf(s0[r] - mnew);
      s1[r] = __expf(s1[r] - mnew);
    }
#pragma unroll
    for (int r = 0; r < 8; ++r) {
      o0[r] *= alpha[r]; o1[r] *= alpha[r];
      o2[r] *= alpha[r]; o3[r] *= alpha[r];
      ol[r] *= alpha[r];
    }

    // C-layout -> A-layout for P via per-wave LDS tile
#pragma unroll
    for (int r = 0; r < 8; ++r) {
      int row = r + 8 * half;
      pl[row * 32 + n]      = (__bf16)s0[r];
      pl[row * 32 + 16 + n] = (__bf16)s1[r];
    }
    v16bf pa = a_frag_bf16(pl, 32, 0, 0, lane);

    // O(16x64) += P(16x32) @ V(32x64): 4 WMMAs, B frags from V^T LDS tile
    // denominator += P @ ones: 1 WMMA, constant B fragment (no loads)
    v16bf vb0 = b_frag_trans_bf16(Vl, 32, 0,  0, lane);
    v16bf vb1 = b_frag_trans_bf16(Vl, 32, 16, 0, lane);
    v16bf vb2 = b_frag_trans_bf16(Vl, 32, 32, 0, lane);
    v16bf vb3 = b_frag_trans_bf16(Vl, 32, 48, 0, lane);
    o0 = wmma_bf16(pa, vb0, o0);
    o1 = wmma_bf16(pa, vb1, o1);
    o2 = wmma_bf16(pa, vb2, o2);
    o3 = wmma_bf16(pa, vb3, o3);
    ol = wmma_bf16(pa, onesb, ol);

    __syncthreads();
  }

  __bf16* Op = O + (size_t)b * SS * HH + h * HDD;
#pragma unroll
  for (int r = 0; r < 8; ++r) {
    int row = q0 + r + 8 * half;
    float inv = 1.0f / ol[r];            // every lane's column holds the row sum
    Op[(size_t)row * HH + 0 + n]  = (__bf16)(o0[r] * inv);
    Op[(size_t)row * HH + 16 + n] = (__bf16)(o1[r] * inv);
    Op[(size_t)row * HH + 32 + n] = (__bf16)(o2[r] * inv);
    Op[(size_t)row * HH + 48 + n] = (__bf16)(o3[r] * inv);
  }
}

// ---------------------------------------------------------------------------
extern "C" void kernel_launch(void* const* d_in, const int* in_sizes, int n_in,
                              void* d_out, int out_size, void* d_ws, size_t ws_size,
                              hipStream_t stream) {
  const float* hs   = (const float*)d_in[0];
  const int*   mask = (const int*)d_in[1];
  const float* Wq = (const float*)d_in[2];
  const float* bq = (const float*)d_in[3];
  const float* Wk = (const float*)d_in[4];
  const float* bk = (const float*)d_in[5];
  const float* Wv = (const float*)d_in[6];
  const float* bv = (const float*)d_in[7];
  const float* Wo = (const float*)d_in[8];
  const float* bo = (const float*)d_in[9];

  const int M = BB * SS, N = HH, K = HH;       // 4096, 1024, 1024
  const size_t mat = (size_t)M * HH;           // 4M elements
  const size_t wmat = (size_t)HH * HH;         // 1M elements

  __bf16* hb  = (__bf16*)d_ws;
  __bf16* qb  = hb + mat;
  __bf16* kb  = qb + mat;
  __bf16* vt  = kb + mat;                      // V^T [B][NH][HD][S]
  __bf16* ab  = vt + mat;
  __bf16* wtq = ab + mat;
  __bf16* wtk = wtq + wmat;
  __bf16* wtv = wtk + wmat;
  __bf16* wto = wtv + wmat;

  // Stage 0: precision staging (fp32 -> bf16; weights also transposed)
  f32_to_bf16_kernel<<<(int)(mat / 4 / 256), 256, 0, stream>>>(hs, hb, (int)mat);
  dim3 tgrid(N / 32, K / 32), tblk(32, 8);
  transpose_f32_bf16_kernel<<<tgrid, tblk, 0, stream>>>(Wq, wtq, K, N);
  transpose_f32_bf16_kernel<<<tgrid, tblk, 0, stream>>>(Wk, wtk, K, N);
  transpose_f32_bf16_kernel<<<tgrid, tblk, 0, stream>>>(Wv, wtv, K, N);
  transpose_f32_bf16_kernel<<<tgrid, tblk, 0, stream>>>(Wo, wto, K, N);

  // Stage 1: QKV projections (V stored head-transposed)
  dim3 ggrid(N / 64, M / 128);                 // 16 x 32 blocks
  gemm_bias_v2<1><<<ggrid, 256, 0, stream>>>(hb, wtq, bq, qb, M, N, K);
  gemm_bias_v2<1><<<ggrid, 256, 0, stream>>>(hb, wtk, bk, kb, M, N, K);
  gemm_bias_v2<2><<<ggrid, 256, 0, stream>>>(hb, wtv, bv, vt, M, N, K);

  // Stage 2: flash attention (block per (b,h,128-query chunk))
  dim3 agrid(SS / 128, BB * NHH);              // 16 x 32
  attention_kernel<<<agrid, 256, 0, stream>>>(qb, kb, vt, mask, ab);

  // Stage 3: output projection (fp32 out)
  gemm_bias_v2<0><<<ggrid, 256, 0, stream>>>(ab, wto, bo, (float*)d_out, M, N, K);
}
